// CustomDecoder_73693048865009
// MI455X (gfx1250) — matrix operand order repaired and checked by reading, not compiled, not run on hardware
//
#include <hip/hip_runtime.h>

// CustomDecoder for MI455X (gfx1250): attention-GRU greedy decoder.
// - enc + weights converted to bf16 once (enc_bf = 128MB -> L2-resident across
//   all 47 steps; 23.3TB/s HBM never re-touched for enc).
// - all small-M GEMMs via v_wmma_f32_16x16x32_bf16; B tiles staged through LDS
//   with global_load_async_to_lds_b128 (ASYNCcnt) + triple buffering so the
//   4 waves of a block fetch each weight tile exactly once.
// - memory-bound attention matvecs split 8-way over S for WGP occupancy.

constexpr int B = 64, S = 1024, H = 1024, V = 1024, T = 48;
constexpr int SOS_IDX = 1;
constexpr int SCHUNK = 128;          // S split for attention kernels

typedef __attribute__((ext_vector_type(16))) __bf16 v16bf;
typedef __attribute__((ext_vector_type(8)))  float  v8f;

static __device__ __forceinline__ unsigned short f2bf(float f) {
    unsigned u = __float_as_uint(f);
    u += 0x7fffu + ((u >> 16) & 1u);          // round-to-nearest-even
    return (unsigned short)(u >> 16);
}
static __device__ __forceinline__ float bf2f(unsigned h) {
    return __uint_as_float(h << 16);
}

union BfFrag { v16bf v; unsigned u[8]; uint4 q[2]; };

// async copy of 16B from global to LDS (per-lane addresses), tracked by ASYNCcnt
static __device__ __forceinline__ void async_lds_b128(unsigned lds_off, const void* gaddr) {
    asm volatile("global_load_async_to_lds_b128 %0, %1, off"
                 :: "v"(lds_off), "v"((unsigned long long)(size_t)gaddr)
                 : "memory");
}

// ---------------------------------------------------------------------------
// f32 -> bf16 bulk convert (4 elems/thread, exact grid)
// ---------------------------------------------------------------------------
__global__ void cvt_f32_bf16_v4(const float4* __restrict__ in, uint2* __restrict__ out) {
    const size_t i = (size_t)blockIdx.x * blockDim.x + threadIdx.x;
    float4 f = in[i];
    uint2 o;
    o.x = (unsigned)f2bf(f.x) | ((unsigned)f2bf(f.y) << 16);
    o.y = (unsigned)f2bf(f.z) | ((unsigned)f2bf(f.w) << 16);
    out[i] = o;
}

// ---------------------------------------------------------------------------
// WMMA GEMM:  C[M=64, N] = A_bf16[64, K] * W_bf16[N, K]^T + bias
// block = (32,4): 4 waves, each owns a 16-row M tile; the block shares one
// 32-column N tile. Per 32-K chunk the 2KB B tile is fetched ONCE via
// global_load_async_to_lds_b128 (128 threads x 16B), triple-buffered in LDS
// (incremental buffer offsets, branch-free steady state, peeled epilogue).
// A/B fragments use contiguous K order (lane half h holds K 16h..16h+15).
// ---------------------------------------------------------------------------
__global__ __launch_bounds__(128)
void gemm_bf16_wmma(const unsigned short* __restrict__ A,
                    const unsigned short* __restrict__ W,
                    const float* __restrict__ bias,
                    float* __restrict__ C,
                    int N, int K)
{
    __shared__ alignas(16) unsigned char sB[3 * 2048];   // 3 x (32 cols x 64B)
    const int tid  = threadIdx.y * 32 + threadIdx.x;
    const int lane = threadIdx.x;          // 0..31 (wave32)
    const int half = lane >> 4;            // 0 | 1
    const int l15  = lane & 15;
    const int m0   = threadIdx.y * 16;     // M tile (M=64 -> 4 tiles)
    const int n0   = blockIdx.x * 32;      // N tile pair

    // B staging assignment: thread -> (column, 16B piece)
    const int bcol = tid >> 2, bpiece = tid & 3;
    const unsigned short* gw = W + (size_t)(n0 + bcol) * K + bpiece * 8;
    const unsigned lbase = (unsigned)(bcol * 64 + bpiece * 16);
    const unsigned sbase = (unsigned)(size_t)&sB[0] + lbase;

    const unsigned short* arow = A + (size_t)(m0 + l15) * K;
    // A dword offsets inside a 32-elem K chunk: lanes 0-15 dwords 0..7
    // (K 0..15); lanes 16-31 dwords 8..15 (K 16..31).
    const int aoff = half * 8;
    // B in LDS: col*64B holds that column's 32 K values; lanes 0-15 read
    // K 0..15 (bytes 0..31), lanes 16-31 read K 16..31 (bytes 32..63).
    const unsigned broff = (unsigned)(half * 32);

    v8f acc0 = {}; v8f acc1 = {};

    // one 32-K chunk of compute, reading the B tile at LDS byte offset rd
    auto compute = [&](const unsigned short* ak, unsigned rd) {
        BfFrag af, bf0, bf1;
        const unsigned* ap = (const unsigned*)ak;
        __builtin_prefetch((const void*)(ak + 128), 0, 1);
#pragma unroll
        for (int i = 0; i < 4; ++i) {
            af.u[i]     = ap[aoff + i];
            af.u[i + 4] = ap[aoff + 4 + i];
        }
        const unsigned char* bufp = sB + rd;
        const uint4* q0 = (const uint4*)(bufp + (unsigned)l15 * 64 + broff);
        const uint4* q1 = (const uint4*)(bufp + (unsigned)(16 + l15) * 64 + broff);
        bf0.q[0] = q0[0]; bf0.q[1] = q0[1];
        bf1.q[0] = q1[0]; bf1.q[1] = q1[1];
        acc0 = __builtin_amdgcn_wmma_f32_16x16x32_bf16(false, af.v, false, bf0.v,
                                                       (short)0, acc0, false, false);
        acc1 = __builtin_amdgcn_wmma_f32_16x16x32_bf16(false, af.v, false, bf1.v,
                                                       (short)0, acc1, false, false);
    };

    const int nk = K / 32;
    unsigned rd = 0, wr = 2048;                       // LDS buffer byte offsets
    const unsigned short* gwk = gw;                   // next chunk to fetch
    const unsigned short* ak  = arow;

    async_lds_b128(sbase, gwk);                       // chunk 0 -> buf 0
    gwk += 32;

    for (int k = 0; k < nk - 1; ++k) {                // steady state: branch-free
        async_lds_b128(sbase + wr, gwk);              // chunk k+1 -> buf wr
        gwk += 32;
        asm volatile("s_wait_asynccnt 0x1" ::: "memory");   // own chunk-k piece landed
        __syncthreads();                                    // all waves' pieces landed
        compute(ak, rd);
        ak += 32;
        rd = (rd == 4096u) ? 0u : rd + 2048u;
        wr = (wr == 4096u) ? 0u : wr + 2048u;
    }
    asm volatile("s_wait_asynccnt 0x0" ::: "memory"); // epilogue: last chunk
    __syncthreads();
    compute(ak, rd);

    // C/D layout: VGPR r -> row m0 + r + half*8, col = l15 (+16 for second tile)
    const int mb = m0 + half * 8;
    const int c0 = n0 + l15, c1 = c0 + 16;
    const float bv0 = bias[c0], bv1 = bias[c1];
#pragma unroll
    for (int r = 0; r < 8; ++r) {
        C[(size_t)(mb + r) * N + c0] = acc0[r] + bv0;
        C[(size_t)(mb + r) * N + c1] = acc1[r] + bv1;
    }
}

// ---------------------------------------------------------------------------
// scores[b,s] = enc[b,s,:].th[b,:]; grid (B, S/SCHUNK) for WGP occupancy.
// Each wave computes one full row-dot per pass (lane covers a contiguous
// 32-element H segment, wave shfl-reduce). enc is bf16 and L2-resident.
// ---------------------------------------------------------------------------
__global__ __launch_bounds__(1024)
void attn_scores_part(const unsigned short* __restrict__ encbf,
                      const float* __restrict__ th,
                      float* __restrict__ scores)
{
    const int b = blockIdx.x;
    const int s0 = blockIdx.y * SCHUNK;
    const int tid = threadIdx.x;
    const int lane = tid & 31, wave = tid >> 5;
    __shared__ float sth[H];

    sth[tid] = th[(size_t)b * H + tid];
    __syncthreads();

    const unsigned short* encb = encbf + (size_t)b * S * H;
    const int hbase = lane * 32;
    for (int pass = 0; pass < SCHUNK / 32; ++pass) {
        const int s = s0 + pass * 32 + wave;
        const unsigned short* rowp = encb + (size_t)s * H + hbase;
        float sum = 0.f;
#pragma unroll
        for (int q = 0; q < 4; ++q) {
            uint4 u = ((const uint4*)rowp)[q];
            const int j = hbase + q * 8;
            sum += bf2f(u.x & 0xffffu) * sth[j + 0];
            sum += bf2f(u.x >> 16)     * sth[j + 1];
            sum += bf2f(u.y & 0xffffu) * sth[j + 2];
            sum += bf2f(u.y >> 16)     * sth[j + 3];
            sum += bf2f(u.z & 0xffffu) * sth[j + 4];
            sum += bf2f(u.z >> 16)     * sth[j + 5];
            sum += bf2f(u.w & 0xffffu) * sth[j + 6];
            sum += bf2f(u.w >> 16)     * sth[j + 7];
        }
#pragma unroll
        for (int off = 16; off > 0; off >>= 1) sum += __shfl_xor(sum, off, 32);
        if (lane == 0) scores[(size_t)b * S + s] = sum;
    }
}

// softmax over S=1024 per batch row (one block per batch)
__global__ __launch_bounds__(1024)
void softmax_row(const float* __restrict__ scores, float* __restrict__ attn)
{
    const int b = blockIdx.x, tid = threadIdx.x;
    const int lane = tid & 31, wave = tid >> 5;
    __shared__ float red[32];

    const float v = scores[(size_t)b * S + tid];
    float m = v;
#pragma unroll
    for (int off = 16; off > 0; off >>= 1) m = fmaxf(m, __shfl_xor(m, off, 32));
    if (lane == 0) red[wave] = m;
    __syncthreads();
    if (wave == 0) {
        float x = red[lane];
#pragma unroll
        for (int off = 16; off > 0; off >>= 1) x = fmaxf(x, __shfl_xor(x, off, 32));
        if (lane == 0) red[0] = x;
    }
    __syncthreads();
    const float gmax = red[0];
    __syncthreads();
    const float e = __expf(v - gmax);
    float s2 = e;
#pragma unroll
    for (int off = 16; off > 0; off >>= 1) s2 += __shfl_xor(s2, off, 32);
    if (lane == 0) red[wave] = s2;
    __syncthreads();
    if (wave == 0) {
        float x = red[lane];
#pragma unroll
        for (int off = 16; off > 0; off >>= 1) x += __shfl_xor(x, off, 32);
        if (lane == 0) red[0] = x;
    }
    __syncthreads();
    attn[(size_t)b * S + tid] = e / red[0];
}

// ---------------------------------------------------------------------------
// ctx partials: ctx_part[b,chunk,h] = sum_{s in chunk} attn[b,s]*enc[b,s,h]
// grid (B, S/SCHUNK); thread = h; fully coalesced column loads from L2.
// ---------------------------------------------------------------------------
__global__ __launch_bounds__(1024)
void attn_ctx_part(const unsigned short* __restrict__ encbf,
                   const float* __restrict__ attn,
                   float* __restrict__ ctx_part)
{
    const int b = blockIdx.x, chunk = blockIdx.y, h = threadIdx.x;
    __shared__ float sa[SCHUNK];
    if (h < SCHUNK) sa[h] = attn[(size_t)b * S + chunk * SCHUNK + h];
    __syncthreads();
    const unsigned short* p = encbf + ((size_t)b * S + (size_t)chunk * SCHUNK) * H + h;
    float acc = 0.f;
    for (int s = 0; s < SCHUNK; s += 4) {
        acc += sa[s]     * bf2f(p[(size_t)(s)     * H]);
        acc += sa[s + 1] * bf2f(p[(size_t)(s + 1) * H]);
        acc += sa[s + 2] * bf2f(p[(size_t)(s + 2) * H]);
        acc += sa[s + 3] * bf2f(p[(size_t)(s + 3) * H]);
    }
    ctx_part[((size_t)b * (S / SCHUNK) + chunk) * H + h] = acc;
}

// deterministic fixed-order reduction of the 8 partials
__global__ void ctx_reduce(const float* __restrict__ part, float* __restrict__ ctx)
{
    const int i = blockIdx.x * blockDim.x + threadIdx.x;   // B*H
    const int b = i >> 10, h = i & (H - 1);
    float s = 0.f;
#pragma unroll
    for (int c = 0; c < S / SCHUNK; ++c)
        s += part[((size_t)b * (S / SCHUNK) + c) * H + h];
    ctx[i] = s;
}

// x = concat(ctx, embed[idx]) in bf16 for the GRU input GEMM
__global__ __launch_bounds__(1024)
void build_x(const float* __restrict__ ctx, const unsigned short* __restrict__ embbf,
             const int* __restrict__ idx, unsigned short* __restrict__ xbf)
{
    const int b = blockIdx.x, h = threadIdx.x;
    xbf[(size_t)b * 2 * H + h]     = f2bf(ctx[(size_t)b * H + h]);
    xbf[(size_t)b * 2 * H + H + h] = embbf[(size_t)idx[b] * H + h];
}

// GRU gate math: h_new = (1-z)*n + z*h
__global__ void gru_update(const float* __restrict__ gi, const float* __restrict__ gh,
                           float* __restrict__ h, unsigned short* __restrict__ hbf)
{
    const int i = blockIdx.x * blockDim.x + threadIdx.x;   // B*H threads
    const int b = i >> 10, c = i & (H - 1);
    const float* gib = gi + (size_t)b * 3 * H;
    const float* ghb = gh + (size_t)b * 3 * H;
    const float ir = gib[c], iz = gib[H + c], inn = gib[2 * H + c];
    const float hr = ghb[c], hz = ghb[H + c], hn  = ghb[2 * H + c];
    const float r = 1.f / (1.f + __expf(-(ir + hr)));
    const float z = 1.f / (1.f + __expf(-(iz + hz)));
    const float n = tanhf(inn + r * hn);
    const float hnew = (1.f - z) * n + z * h[i];
    h[i]  = hnew;
    hbf[i] = f2bf(hnew);
}

// argmax over V per batch (ties -> lowest index, as jnp.argmax)
__global__ __launch_bounds__(1024)
void argmax_kernel(const float* __restrict__ logits, int* __restrict__ idx,
                   float* __restrict__ out, int t)
{
    const int b = blockIdx.x, tid = threadIdx.x;
    __shared__ float sv[V];
    __shared__ int   si[V];
    sv[tid] = logits[(size_t)b * V + tid];
    si[tid] = tid;
    __syncthreads();
    for (int off = 512; off > 0; off >>= 1) {
        if (tid < off) {
            const float vo = sv[tid + off]; const int io = si[tid + off];
            if (vo > sv[tid] || (vo == sv[tid] && io < si[tid])) { sv[tid] = vo; si[tid] = io; }
        }
        __syncthreads();
    }
    if (tid == 0) {
        idx[b] = si[0];
        out[(size_t)B * T * V + (size_t)b * T + t] = (float)si[0];
    }
}

// decoder_outputs[b, t, :] = embed_w[idx[b], :]  (f32 originals)
__global__ __launch_bounds__(1024)
void write_emb(const float* __restrict__ embed_w, const int* __restrict__ idx,
               float* __restrict__ out, int t)
{
    const int b = blockIdx.x, v = threadIdx.x;
    out[((size_t)b * T + t) * V + v] = embed_w[(size_t)idx[b] * H + v];
}

// h0 = 0, idx0 = SOS, step-0 outputs (one-hot SOS + sampled idx)
__global__ void init_state(float* __restrict__ h, unsigned short* __restrict__ hbf,
                           int* __restrict__ idx, float* __restrict__ out)
{
    const int i = blockIdx.x * blockDim.x + threadIdx.x;   // B*H threads
    h[i] = 0.f; hbf[i] = 0;
    const int b = i >> 10, v = i & (H - 1);
    out[(size_t)b * T * V + v] = (v == SOS_IDX) ? 1.f : 0.f;
    if (i < B) {
        idx[i] = SOS_IDX;
        out[(size_t)B * T * V + (size_t)i * T] = (float)SOS_IDX;
    }
}

// ---------------------------------------------------------------------------
extern "C" void kernel_launch(void* const* d_in, const int* in_sizes, int n_in,
                              void* d_out, int out_size, void* d_ws, size_t ws_size,
                              hipStream_t stream)
{
    const float* enc    = (const float*)d_in[0];
    const float* emb_w  = (const float*)d_in[1];
    const float* attn_w = (const float*)d_in[2];
    const float* attn_b = (const float*)d_in[3];
    const float* w_ih   = (const float*)d_in[4];
    const float* w_hh   = (const float*)d_in[5];
    const float* b_ih   = (const float*)d_in[6];
    const float* b_hh   = (const float*)d_in[7];
    const float* out_w  = (const float*)d_in[8];
    const float* out_b  = (const float*)d_in[9];
    float* out = (float*)d_out;

    char* p = (char*)d_ws;
    auto alloc = [&](size_t bytes) -> char* {
        char* r = p; p += (bytes + 255) & ~(size_t)255; return r;
    };
    unsigned short* enc_bf = (unsigned short*)alloc((size_t)B * S * H * 2);  // 128 MB (L2-resident)
    unsigned short* emb_bf = (unsigned short*)alloc((size_t)V * H * 2);
    unsigned short* aw_bf  = (unsigned short*)alloc((size_t)H * H * 2);
    unsigned short* wih_bf = (unsigned short*)alloc((size_t)3 * H * 2 * H * 2);
    unsigned short* whh_bf = (unsigned short*)alloc((size_t)3 * H * H * 2);
    unsigned short* ow_bf  = (unsigned short*)alloc((size_t)V * H * 2);
    float* h       = (float*)alloc((size_t)B * H * 4);
    unsigned short* h_bf = (unsigned short*)alloc((size_t)B * H * 2);
    float* th      = (float*)alloc((size_t)B * H * 4);
    float* scores  = (float*)alloc((size_t)B * S * 4);
    float* attn    = (float*)alloc((size_t)B * S * 4);
    float* ctx_prt = (float*)alloc((size_t)B * (S / SCHUNK) * H * 4);
    float* ctx     = (float*)alloc((size_t)B * H * 4);
    unsigned short* x_bf = (unsigned short*)alloc((size_t)B * 2 * H * 2);
    float* gi      = (float*)alloc((size_t)B * 3 * H * 4);
    float* gh      = (float*)alloc((size_t)B * 3 * H * 4);
    float* logits  = (float*)alloc((size_t)B * V * 4);
    int* idx       = (int*)alloc((size_t)B * 4);

    auto cvt = [&](const float* src, unsigned short* dst, size_t n) {
        cvt_f32_bf16_v4<<<dim3((unsigned)(n / 4 / 256)), dim3(256), 0, stream>>>(
            (const float4*)src, (uint2*)dst);
    };
    cvt(enc,    enc_bf, (size_t)B * S * H);
    cvt(emb_w,  emb_bf, (size_t)V * H);
    cvt(attn_w, aw_bf,  (size_t)H * H);
    cvt(w_ih,   wih_bf, (size_t)3 * H * 2 * H);
    cvt(w_hh,   whh_bf, (size_t)3 * H * H);
    cvt(out_w,  ow_bf,  (size_t)V * H);

    init_state<<<dim3(B * H / 256), dim3(256), 0, stream>>>(h, h_bf, idx, out);

    const dim3 gblk(32, 4);
    for (int t = 1; t < T; ++t) {
        // th = h @ attn_w^T + attn_b
        gemm_bf16_wmma<<<dim3(H / 32), gblk, 0, stream>>>(h_bf, aw_bf, attn_b, th, H, H);
        // attention over L2-resident bf16 enc (split over S for occupancy)
        attn_scores_part<<<dim3(B, S / SCHUNK), dim3(1024), 0, stream>>>(enc_bf, th, scores);
        softmax_row<<<dim3(B), dim3(1024), 0, stream>>>(scores, attn);
        attn_ctx_part<<<dim3(B, S / SCHUNK), dim3(1024), 0, stream>>>(enc_bf, attn, ctx_prt);
        ctx_reduce<<<dim3(B * H / 256), dim3(256), 0, stream>>>(ctx_prt, ctx);
        // x = [ctx, emb[idx]]
        build_x<<<dim3(B), dim3(1024), 0, stream>>>(ctx, emb_bf, idx, x_bf);
        // gi = x @ W_ih^T + b_ih ; gh = h @ W_hh^T + b_hh
        gemm_bf16_wmma<<<dim3(3 * H / 32), gblk, 0, stream>>>(x_bf, wih_bf, b_ih, gi, 3 * H, 2 * H);
        gemm_bf16_wmma<<<dim3(3 * H / 32), gblk, 0, stream>>>(h_bf, whh_bf, b_hh, gh, 3 * H, H);
        gru_update<<<dim3(B * H / 256), dim3(256), 0, stream>>>(gi, gh, h, h_bf);
        // logits = h_new @ out_w^T + out_b ; greedy argmax; emit outputs
        gemm_bf16_wmma<<<dim3(V / 32), gblk, 0, stream>>>(h_bf, ow_bf, out_b, logits, V, H);
        argmax_kernel<<<dim3(B), dim3(1024), 0, stream>>>(logits, idx, out, t);
        write_emb<<<dim3(B), dim3(1024), 0, stream>>>(emb_w, idx, out, t);
    }
    (void)in_sizes; (void)n_in; (void)out_size; (void)ws_size;
}
